// CrossAttentionLayer_39582418600215
// MI455X (gfx1250) — matrix operand order repaired and checked by reading, not compiled, not run on hardware
//
#include <hip/hip_runtime.h>
#include <hip/hip_bf16.h>

// ---------------------------------------------------------------------------
// CDNA5 (gfx1250) cross-attention layer.
// All heavy math goes through v_wmma_f32_16x16x32_bf16 (wave32 WMMA).
// GEMM: 64x128 block tile, K-step 32, double-buffered LDS, 8 wave32/block,
// each wave computes a 32x32 register tile = 4 WMMA accumulators.
// ---------------------------------------------------------------------------

typedef __attribute__((ext_vector_type(16))) __bf16 v16bf;
typedef __attribute__((ext_vector_type(8)))  float  v8f;

union FragV {
    v16bf v;
    uint4 q[2];
};

__device__ __forceinline__ unsigned short f2bf(float f) {
    // round-to-nearest-even fp32 -> bf16
    unsigned int u = __float_as_uint(f);
    u += 0x7FFFu + ((u >> 16) & 1u);
    return (unsigned short)(u >> 16);
}

__device__ __forceinline__ uint4 pack8(float4 f0, float4 f1) {
    uint4 p;
    p.x = (unsigned)f2bf(f0.x) | ((unsigned)f2bf(f0.y) << 16);
    p.y = (unsigned)f2bf(f0.z) | ((unsigned)f2bf(f0.w) << 16);
    p.z = (unsigned)f2bf(f1.x) | ((unsigned)f2bf(f1.y) << 16);
    p.w = (unsigned)f2bf(f1.z) | ((unsigned)f2bf(f1.w) << 16);
    return p;
}

#define EPSN 1e-8f
#define SMOOTH_F 9.0f

// ---------------------------------------------------------------------------
// Generic batched tiled GEMM:  C[b] = A[b] (MxK) * B[b]  (+ epilogue)
//   A_BF16 : A operand stored as bf16 (ushort) instead of fp32
//   B_KN   : B stored [K][N] (NN, transpose-staged into LDS); else [N][K] (NT)
//   EPI    : 0 = relu -> f32 store ; 1 = bf16 store ; 2 = bias + relu -> f32
// ---------------------------------------------------------------------------
template <bool A_BF16, bool B_KN, int EPI>
__global__ __launch_bounds__(256) void gemm_wmma_k(
    const void* __restrict__ Ap, const float* __restrict__ Bp,
    const float* __restrict__ bias, void* __restrict__ Cp,
    int M, int N, int K,
    long long strA, long long strB, long long strC)
{
    // double-buffered bf16 tiles, LDS layout [row][K] (row = M for A, N for B)
    __shared__ unsigned short sA[2][64 * 32];
    __shared__ unsigned short sB[2][128 * 32];

    const int t  = threadIdx.x;
    const int tN = blockIdx.x * 128;
    const int tM = blockIdx.y * 64;
    const int b  = blockIdx.z;

    const float*          Af = nullptr;
    const unsigned short* Ah = nullptr;
    if (A_BF16) Ah = (const unsigned short*)Ap + (size_t)b * strA;
    else        Af = (const float*)Ap + (size_t)b * strA;
    const float* Bf = Bp + (size_t)b * strB;

    const int wave = t >> 5;
    const int lane = t & 31;
    const int mf0  = (wave & 1) * 2;  // first M-fragment (0 or 2)
    const int nf0  = (wave >> 1) * 2; // first N-fragment (0,2,4,6)

    v8f acc00 = {}, acc01 = {}, acc10 = {}, acc11 = {};

    // staging indices (A: 8 elems/thread, B: 16 elems/thread)
    const int ar = t >> 2;          // 0..63   A row
    const int ac = (t & 3) * 8;     // 0..24   A K-offset
    const int br = t >> 1;          // 0..127  B row (NT)
    const int bc = (t & 1) * 16;    // 0/16    B K-offset (NT)
    const int kr = t >> 3;          // 0..31   B K-row (NN)
    const int nc = (t & 7) * 16;    // 0..112  B N-offset (NN)

    // ---- global load (+ bf16 convert) for the tile starting at k0 --------
    auto loadA = [&](int k0) -> uint4 {
        if (A_BF16) {
            return *(const uint4*)(Ah + (size_t)(tM + ar) * K + k0 + ac);
        } else {
            const float* g = Af + (size_t)(tM + ar) * K + k0 + ac;
            return pack8(*(const float4*)g, *(const float4*)(g + 4));
        }
    };
    auto loadB_NT = [&](int k0, uint4& p0, uint4& p1) {
        const float* g = Bf + (size_t)(tN + br) * K + k0 + bc;
        p0 = pack8(*(const float4*)g,       *(const float4*)(g + 4));
        p1 = pack8(*(const float4*)(g + 8), *(const float4*)(g + 12));
    };
    auto loadB_NN = [&](int k0, unsigned short* us) {
        const float* g = Bf + (size_t)(k0 + kr) * N + tN + nc;
        float4 f0 = *(const float4*)g;
        float4 f1 = *(const float4*)(g + 4);
        float4 f2 = *(const float4*)(g + 8);
        float4 f3 = *(const float4*)(g + 12);
        float fv[16] = {f0.x, f0.y, f0.z, f0.w, f1.x, f1.y, f1.z, f1.w,
                        f2.x, f2.y, f2.z, f2.w, f3.x, f3.y, f3.z, f3.w};
#pragma unroll
        for (int j = 0; j < 16; ++j) us[j] = f2bf(fv[j]);
    };

    // ---- LDS stores ------------------------------------------------------
    auto storeA = [&](int buf, uint4 p) {
        *(uint4*)&sA[buf][ar * 32 + ac] = p;
    };
    auto storeB_NT = [&](int buf, uint4 p0, uint4 p1) {
        *(uint4*)&sB[buf][br * 32 + bc]     = p0;
        *(uint4*)&sB[buf][br * 32 + bc + 8] = p1;
    };
    auto storeB_NN = [&](int buf, const unsigned short* us) {
#pragma unroll
        for (int j = 0; j < 16; ++j)
            sB[buf][(nc + j) * 32 + kr] = us[j]; // transpose into [N][K]
    };

    // ---- WMMA compute from LDS buffer ------------------------------------
    auto compute = [&](int buf) {
        FragV a0, a1, bb0, bb1;
        // A 16x32 bf16: lane<16 -> M=lane, K 0..15;
        //               lane>=16 -> M=lane-16, K 16..31
        const int kha = (lane >> 4) * 16;
        const int rA0 = (mf0 * 16 + (lane & 15)) * 32;
        const int rA1 = ((mf0 + 1) * 16 + (lane & 15)) * 32;
        a0.q[0] = *(const uint4*)&sA[buf][rA0 + kha];
        a0.q[1] = *(const uint4*)&sA[buf][rA0 + kha + 8];
        a1.q[0] = *(const uint4*)&sA[buf][rA1 + kha];
        a1.q[1] = *(const uint4*)&sA[buf][rA1 + kha + 8];
        // B 32x16 bf16: lane holds column N=lane%16, 16 contiguous K (lane/16)
        const int khb = (lane >> 4) * 16;
        const int rB0 = (nf0 * 16 + (lane & 15)) * 32 + khb;
        const int rB1 = ((nf0 + 1) * 16 + (lane & 15)) * 32 + khb;
        bb0.q[0] = *(const uint4*)&sB[buf][rB0];
        bb0.q[1] = *(const uint4*)&sB[buf][rB0 + 8];
        bb1.q[0] = *(const uint4*)&sB[buf][rB1];
        bb1.q[1] = *(const uint4*)&sB[buf][rB1 + 8];

        acc00 = __builtin_amdgcn_wmma_f32_16x16x32_bf16(
            false, a0.v, false, bb0.v, (short)0, acc00, false, false);
        acc10 = __builtin_amdgcn_wmma_f32_16x16x32_bf16(
            false, a1.v, false, bb0.v, (short)0, acc10, false, false);
        acc01 = __builtin_amdgcn_wmma_f32_16x16x32_bf16(
            false, a0.v, false, bb1.v, (short)0, acc01, false, false);
        acc11 = __builtin_amdgcn_wmma_f32_16x16x32_bf16(
            false, a1.v, false, bb1.v, (short)0, acc11, false, false);
    };

    // ---- prologue: stage tile 0 ------------------------------------------
    uint4 apk = loadA(0);
    uint4 bp0, bp1;
    unsigned short bus[16];
    if (!B_KN) loadB_NT(0, bp0, bp1);
    else       loadB_NN(0, bus);
    storeA(0, apk);
    if (!B_KN) storeB_NT(0, bp0, bp1);
    else       storeB_NN(0, bus);
    __syncthreads();

    // ---- main loop: global loads for tile i+1 overlap compute on tile i --
    int cur = 0;
    for (int k0 = 0; k0 < K; k0 += 32) {
        const bool last = (k0 + 32 >= K);
        if (!last) {
            apk = loadA(k0 + 32);
            if (!B_KN) loadB_NT(k0 + 32, bp0, bp1);
            else       loadB_NN(k0 + 32, bus);
        }
        compute(cur);
        if (!last) {
            storeA(cur ^ 1, apk);
            if (!B_KN) storeB_NT(cur ^ 1, bp0, bp1);
            else       storeB_NN(cur ^ 1, bus);
            __syncthreads();
            cur ^= 1;
        }
    }

    // ---- epilogue --------------------------------------------------------
    // C/D f32 layout: VGPR r holds M = r + 8*(lane>=16), N = lane%16
    const int rsel = (lane >> 4) << 3;
    const int row0 = tM + mf0 * 16 + rsel;
    const int row1 = tM + (mf0 + 1) * 16 + rsel;
    const int col0 = tN + nf0 * 16 + (lane & 15);
    const int col1 = tN + (nf0 + 1) * 16 + (lane & 15);

    auto store_frag = [&](const v8f& a, int row, int col) {
        if (EPI == 0) {
            float* C = (float*)Cp + (size_t)b * strC;
#pragma unroll
            for (int r = 0; r < 8; ++r)
                C[(size_t)(row + r) * N + col] = fmaxf(a[r], 0.0f);
        } else if (EPI == 1) {
            unsigned short* C = (unsigned short*)Cp + (size_t)b * strC;
#pragma unroll
            for (int r = 0; r < 8; ++r)
                C[(size_t)(row + r) * N + col] = f2bf(a[r]);
        } else {
            float* C = (float*)Cp + (size_t)b * strC;
            const float bv = bias[col];
#pragma unroll
            for (int r = 0; r < 8; ++r)
                C[(size_t)(row + r) * N + col] = fmaxf(a[r] + bv, 0.0f);
        }
    };
    store_frag(acc00, row0, col0);
    store_frag(acc10, row1, col0);
    store_frag(acc01, row0, col1);
    store_frag(acc11, row1, col1);
}

// ---------------------------------------------------------------------------
// denom1[b,c] = sum_q R[b,c,q]   (row sums over Lt=256 contiguous floats)
// ---------------------------------------------------------------------------
__global__ __launch_bounds__(256) void rowsum_k(const float* __restrict__ R,
                                                float* __restrict__ d1)
{
    __shared__ float sred[256];
    const int c = blockIdx.x, b = blockIdx.y, t = threadIdx.x;
    const float* row = R + ((size_t)b * 576 + c) * 256;
    sred[t] = row[t];
    __syncthreads();
    for (int s = 128; s > 0; s >>= 1) {
        if (t < s) sred[t] += sred[t + s];
        __syncthreads();
    }
    if (t == 0) d1[(size_t)b * 576 + c] = sred[0];
}

// ---------------------------------------------------------------------------
// denom2[b,q] = sum_c R[b,c,q]   (column sums; coalesced across threads)
// ---------------------------------------------------------------------------
__global__ __launch_bounds__(256) void colsum_k(const float* __restrict__ R,
                                                float* __restrict__ d2)
{
    const int b = blockIdx.x, q = threadIdx.x;
    const float* Rb = R + (size_t)b * 576 * 256;
    float s = 0.0f;
    for (int c = 0; c < 576; ++c) s += Rb[(size_t)c * 256 + q];
    d2[(size_t)b * 256 + q] = s;
}

// ---------------------------------------------------------------------------
// attn_img[b,q,c] = softmax_c( R[b,c,q] / (denom1[b,c]+eps) * 9 ),  c < 576
// one block per (q txt-token, b); R is L2-resident (75 MB < 192 MB L2)
// ---------------------------------------------------------------------------
__global__ __launch_bounds__(256) void attn_img_k(const float* __restrict__ R,
                                                  const float* __restrict__ d1,
                                                  float* __restrict__ out)
{
    __shared__ float sred[256];
    const int q = blockIdx.x, b = blockIdx.y, t = threadIdx.x;
    const float* Rb = R + (size_t)b * 576 * 256;
    const float* db = d1 + (size_t)b * 576;

    float v[3];
    int n = 0;
    float mx = -1e30f;
    for (int c = t; c < 576; c += 256) {
        float x = Rb[(size_t)c * 256 + q] / (db[c] + EPSN) * SMOOTH_F;
        v[n++] = x;
        mx = fmaxf(mx, x);
    }
    sred[t] = mx;
    __syncthreads();
    for (int s = 128; s > 0; s >>= 1) {
        if (t < s) sred[t] = fmaxf(sred[t], sred[t + s]);
        __syncthreads();
    }
    mx = sred[0];
    __syncthreads();

    float sum = 0.0f;
    for (int i = 0; i < n; ++i) {
        v[i] = __expf(v[i] - mx);
        sum += v[i];
    }
    sred[t] = sum;
    __syncthreads();
    for (int s = 128; s > 0; s >>= 1) {
        if (t < s) sred[t] += sred[t + s];
        __syncthreads();
    }
    sum = sred[0];

    float* ob = out + ((size_t)b * 256 + q) * 576;
    n = 0;
    for (int c = t; c < 576; c += 256) ob[c] = v[n++] / sum;
}

// ---------------------------------------------------------------------------
// attn_txt[b,q,c] = softmax_c( R[b,q,c] / (denom2[b,c]+eps) * 9 ),  c < 256
// one block per (q img-region, b); fully contiguous
// ---------------------------------------------------------------------------
__global__ __launch_bounds__(256) void attn_txt_k(const float* __restrict__ R,
                                                  const float* __restrict__ d2,
                                                  float* __restrict__ out)
{
    __shared__ float sred[256];
    const int q = blockIdx.x, b = blockIdx.y, t = threadIdx.x;
    float x = R[((size_t)b * 576 + q) * 256 + t] /
              (d2[(size_t)b * 256 + t] + EPSN) * SMOOTH_F;
    sred[t] = x;
    __syncthreads();
    for (int s = 128; s > 0; s >>= 1) {
        if (t < s) sred[t] = fmaxf(sred[t], sred[t + s]);
        __syncthreads();
    }
    const float mx = sred[0];
    __syncthreads();
    const float e = __expf(x - mx);
    sred[t] = e;
    __syncthreads();
    for (int s = 128; s > 0; s >>= 1) {
        if (t < s) sred[t] += sred[t + s];
        __syncthreads();
    }
    out[((size_t)b * 576 + q) * 256 + t] = e / sred[0];
}

// ---------------------------------------------------------------------------
// Launch: score GEMM -> L1 denominators -> softmaxes (into d_out) ->
// weighted-context GEMMs (bf16 ws) -> FC GEMMs (bias+relu, into d_out)
// ---------------------------------------------------------------------------
extern "C" void kernel_launch(void* const* d_in, const int* in_sizes, int n_in,
                              void* d_out, int out_size, void* d_ws, size_t ws_size,
                              hipStream_t stream)
{
    (void)in_sizes; (void)n_in; (void)out_size; (void)ws_size;

    const float* txt = (const float*)d_in[0]; // [128,256,1024]
    const float* img = (const float*)d_in[1]; // [128,576,1024]
    const float* Wt  = (const float*)d_in[2]; // [1024,1024]
    const float* bt  = (const float*)d_in[3]; // [1024]
    const float* Wi  = (const float*)d_in[4]; // [1024,1024]
    const float* bi  = (const float*)d_in[5]; // [1024]

    const int Bn = 128, Lt = 256, Li = 576, H = 1024;

    // workspace layout
    float* R      = (float*)d_ws;                  // relu(img @ txt^T): 18,874,368 f32
    float* denom1 = R + 18874368;                  // [B,576]
    float* denom2 = denom1 + 73728;                // [B,256]
    unsigned short* w1 = (unsigned short*)(denom2 + 32768); // txt_attn_embed bf16
    unsigned short* w2 = w1 + 33554432;            // img_attn_embed bf16

    // output layout (concatenated in return order)
    float* txt_out  = (float*)d_out;               // [B,256,1024]
    float* img_out  = txt_out + 33554432;          // [B,576,1024]
    float* attn_img = txt_out + 109051904;         // [B,256,576]
    float* attn_txt = txt_out + 127926272;         // [B,576,256]

    const dim3 blk(256);

    // 1) R = relu( img[b] @ txt[b]^T )   M=576 N=256 K=1024 (NT)
    gemm_wmma_k<false, false, 0><<<dim3(Lt / 128, Li / 64, Bn), blk, 0, stream>>>(
        img, txt, nullptr, R, Li, Lt, H,
        (long long)Li * H, (long long)Lt * H, (long long)Li * Lt);

    // 2) L1 denominators
    rowsum_k<<<dim3(Li, Bn), blk, 0, stream>>>(R, denom1);
    colsum_k<<<dim3(Bn), blk, 0, stream>>>(R, denom2);

    // 3) smoothed softmaxes (attention outputs, written straight into d_out)
    attn_img_k<<<dim3(Lt, Bn), blk, 0, stream>>>(R, denom1, attn_img);
    attn_txt_k<<<dim3(Li, Bn), blk, 0, stream>>>(R, denom2, attn_txt);

    // 4) weighted contexts (NN: B stored [K][N]) -> bf16 workspace
    //    txt_attn_embed = attn_img @ img : M=256 N=1024 K=576
    gemm_wmma_k<false, true, 1><<<dim3(H / 128, Lt / 64, Bn), blk, 0, stream>>>(
        attn_img, img, nullptr, w1, Lt, H, Li,
        (long long)Lt * Li, (long long)Li * H, (long long)Lt * H);
    //    img_attn_embed = attn_txt @ txt : M=576 N=1024 K=256
    gemm_wmma_k<false, true, 1><<<dim3(H / 128, Li / 64, Bn), blk, 0, stream>>>(
        attn_txt, txt, nullptr, w2, Li, H, Lt,
        (long long)Li * Lt, (long long)Lt * H, (long long)Li * H);

    // 5) FC layers (NT against W, bias + relu) -> d_out
    gemm_wmma_k<true, false, 2><<<dim3(H / 128, Lt / 64, Bn), blk, 0, stream>>>(
        w1, Wt, bt, txt_out, Lt, H, H,
        (long long)Lt * H, 0LL, (long long)Lt * H);
    gemm_wmma_k<true, false, 2><<<dim3(H / 128, Li / 64, Bn), blk, 0, stream>>>(
        w2, Wi, bi, img_out, Li, H, H,
        (long long)Li * H, 0LL, (long long)Li * H);
}
